// Net_85555748537064
// MI455X (gfx1250) — matrix operand order repaired and checked by reading, not compiled, not run on hardware
//
#include <hip/hip_runtime.h>
#include <hip/hip_bf16.h>

// ---------------- problem constants (match reference) ----------------
#define NNODES 100000
#define FIN    512
#define NPATH  1600000
#define H1_    8
#define C1_    8
#define HC1    64     // H1*C1
#define H2_    1
#define C2_    7
#define NEG_SLOPE 0.2f

typedef __attribute__((ext_vector_type(16))) __bf16 bf16x16;
typedef __attribute__((ext_vector_type(8)))  float  f32x8;

// ---------------- helpers ----------------
__device__ __forceinline__ void atomic_max_f32(float* addr, float v) {
  // ordered-int trick; valid when addr initialized to -inf
  if (v >= 0.0f) {
    atomicMax((int*)addr, __float_as_int(v));
  } else {
    atomicMin((unsigned int*)addr, __float_as_uint(v));
  }
}

__global__ void fill_f32(float* __restrict__ p, size_t n, float v) {
  size_t i = (size_t)blockIdx.x * blockDim.x + threadIdx.x;
  if (i < n) p[i] = v;
}

// ---------------- B pre-pack: f32 [K,NC] -> bf16 WMMA fragments ----------------
// Fragment layout: frag (t, w) holds B rows k = t*32 + (j<8 ? half*8 + j : 16 + half*8 + j-8), col = w*16 + l16,
// stored as Bp[((t*NTILES + w)*32 + lane)*16 + j]. Zero-padded for col >= NC.
template <int KSTEPS, int NTILES>
__global__ void pack_B(const float* __restrict__ W, __bf16* __restrict__ Bp, int NC) {
  int idx = blockIdx.x * blockDim.x + threadIdx.x;
  constexpr int total = KSTEPS * NTILES * 32;
  if (idx >= total) return;
  int lane = idx & 31;
  int w    = (idx >> 5) % NTILES;
  int t    = (idx >> 5) / NTILES;
  int half = lane >> 4;
  int l16  = lane & 15;
  int n    = w * 16 + l16;
  __bf16* o = Bp + (size_t)idx * 16;
#pragma unroll
  for (int j = 0; j < 16; ++j) {
    int kk = t * 32 + (j < 8 ? half * 8 + j : 16 + half * 8 + (j - 8));
    float v = (n < NC) ? W[(size_t)kk * NC + n] : 0.0f;
    o[j] = (__bf16)v;
  }
}

// ---------------- WMMA bf16 GEMM: out[n,NC] = A[n,K] @ B[K,NC] ----------------
// One wave owns one 16-row M tile and accumulates all NTILES column tiles,
// so A is read exactly once. No lane-conditional code in the K loop: A loads
// are unconditional float4 (rows always valid: grid covers ceil(nrows/16)
// tiles with whole-wave early exit), B fragments are single 32B vector loads
// from the pre-packed bf16 buffer. EXEC is all-ones at every v_wmma.
template <int KSTEPS, int NTILES>
__global__ void wmma_gemm_bf16(const float* __restrict__ A,
                               const __bf16* __restrict__ Bp,
                               float* __restrict__ out,
                               int nrows, int NC) {
  constexpr int K = KSTEPS * 32;
  const int lane  = threadIdx.x & 31;
  const int wave  = threadIdx.x >> 5;
  const int mtile = blockIdx.x * (blockDim.x >> 5) + wave;
  const int m0    = mtile * 16;
  if (m0 >= nrows) return;          // uniform per wave
  const int half = lane >> 4;       // 0: lanes 0-15, 1: lanes 16-31
  const int l16  = lane & 15;
  const int m    = m0 + l16;        // always < nrows (nrows % 16 == 0 here)

  f32x8 acc[NTILES] = {};
  const float* arow = A + (size_t)m * K;

#pragma unroll
  for (int t = 0; t < KSTEPS; ++t) {
    const int k0 = t * 32;
    // 16-bit A 16x32 layout (ISA p.108): lanes 0-15 K=0..7,16..23; lanes 16-31 K=8..15,24..31
    const float4* alo = (const float4*)(arow + k0 + half * 8);
    const float4* ahi = (const float4*)(arow + k0 + 16 + half * 8);
    float4 p0 = alo[0], p1 = alo[1], p2 = ahi[0], p3 = ahi[1];
    bf16x16 a;
    a[0]  = (__bf16)p0.x; a[1]  = (__bf16)p0.y; a[2]  = (__bf16)p0.z; a[3]  = (__bf16)p0.w;
    a[4]  = (__bf16)p1.x; a[5]  = (__bf16)p1.y; a[6]  = (__bf16)p1.z; a[7]  = (__bf16)p1.w;
    a[8]  = (__bf16)p2.x; a[9]  = (__bf16)p2.y; a[10] = (__bf16)p2.z; a[11] = (__bf16)p2.w;
    a[12] = (__bf16)p3.x; a[13] = (__bf16)p3.y; a[14] = (__bf16)p3.z; a[15] = (__bf16)p3.w;
#pragma unroll
    for (int w = 0; w < NTILES; ++w) {
      bf16x16 b = *(const bf16x16*)(Bp + ((size_t)(t * NTILES + w) * 32 + lane) * 16);
      acc[w] = __builtin_amdgcn_wmma_f32_16x16x32_bf16(
          /*neg_a=*/false, a, /*neg_b=*/false, b,
          /*c_mod=*/(short)0, acc[w], /*reuse_a=*/false, /*reuse_b=*/false);
    }
  }

  // C/D 16x16 f32 layout: vgpr j -> row m0 + j + half*8, lane l16 -> col
#pragma unroll
  for (int w = 0; w < NTILES; ++w) {
    int n = w * 16 + l16;
    if (n < NC) {
#pragma unroll
      for (int j = 0; j < 8; ++j) {
        out[(size_t)(m0 + j + half * 8) * NC + n] = acc[w][j];
      }
    }
  }
}

// ---------------- path kernels ----------------
template <int HC>
__device__ __forceinline__ void load_row(const float* __restrict__ src, float* dst) {
  if constexpr ((HC % 4) == 0) {
#pragma unroll
    for (int i = 0; i < HC / 4; ++i) ((float4*)dst)[i] = ((const float4*)src)[i];
  } else {
#pragma unroll
    for (int j = 0; j < HC; ++j) dst[j] = src[j];
  }
}

// e[p,h] = leaky_relu( xp[src]·att0 + xp[mid]·att1 + xp[dst]·att2 ), fused
// segment-max into mx[dst,h].
template <int H, int C>
__global__ void path_logits_max(const float* __restrict__ xp,
                                const long long* __restrict__ pi,
                                const float* __restrict__ att,   // [3,H,C]
                                float* __restrict__ e,           // [P,H]
                                float* __restrict__ mx) {        // [N,H]
  long long p = (long long)blockIdx.x * blockDim.x + threadIdx.x;
  if (p >= NPATH) return;
  long long s = pi[p];
  long long m = pi[(long long)NPATH + p];
  long long d = pi[2LL * NPATH + p];
  constexpr int HC = H * C;
  float xs[HC], xm[HC], xd[HC];
  load_row<HC>(xp + s * HC, xs);
  load_row<HC>(xp + m * HC, xm);
  load_row<HC>(xp + d * HC, xd);
#pragma unroll
  for (int h = 0; h < H; ++h) {
    float acc = 0.0f;
#pragma unroll
    for (int c = 0; c < C; ++c) {
      int j = h * C + c;
      acc += xs[j] * att[j] + xm[j] * att[HC + j] + xd[j] * att[2 * HC + j];
    }
    acc = (acc > 0.0f) ? acc : NEG_SLOPE * acc;
    e[p * H + h] = acc;
    atomic_max_f32(&mx[d * H + h], acc);
  }
}

// e[p,h] <- exp(e - mx[dst,h]); den[dst,h] += e
template <int H>
__global__ void path_exp_sum(const long long* __restrict__ pi,
                             const float* __restrict__ mx,
                             float* __restrict__ e,
                             float* __restrict__ den) {
  long long p = (long long)blockIdx.x * blockDim.x + threadIdx.x;
  if (p >= NPATH) return;
  long long d = pi[2LL * NPATH + p];
#pragma unroll
  for (int h = 0; h < H; ++h) {
    float v = __expf(e[p * H + h] - mx[d * H + h]);
    e[p * H + h] = v;
    atomicAdd(&den[d * H + h], v);
  }
}

// agg[dst] += (e / (den[dst]+eps)) * xp[src]
template <int H, int C>
__global__ void path_scatter(const float* __restrict__ xp,
                             const long long* __restrict__ pi,
                             const float* __restrict__ e,
                             const float* __restrict__ den,
                             float* __restrict__ agg) {
  long long p = (long long)blockIdx.x * blockDim.x + threadIdx.x;
  if (p >= NPATH) return;
  long long s = pi[p];
  long long d = pi[2LL * NPATH + p];
  constexpr int HC = H * C;
  float xs[HC];
  load_row<HC>(xp + s * HC, xs);
  float* od = agg + d * HC;
#pragma unroll
  for (int h = 0; h < H; ++h) {
    float alpha = e[p * H + h] / (den[d * H + h] + 1e-16f);
#pragma unroll
    for (int c = 0; c < C; ++c) {
      atomicAdd(&od[h * C + c], alpha * xs[h * C + c]);
    }
  }
}

// in-place: a = elu(a + bias[col])
__global__ void bias_elu(float* __restrict__ a, const float* __restrict__ bias,
                         int ncols, long long total) {
  long long i = (long long)blockIdx.x * blockDim.x + threadIdx.x;
  if (i >= total) return;
  float v = a[i] + bias[(int)(i % ncols)];
  a[i] = (v > 0.0f) ? v : (__expf(v) - 1.0f);
}

// out[n,:] = log_softmax(agg[n,:] + bias), 7 classes
__global__ void bias_logsoftmax7(const float* __restrict__ agg,
                                 const float* __restrict__ bias,
                                 float* __restrict__ out) {
  int n = blockIdx.x * blockDim.x + threadIdx.x;
  if (n >= NNODES) return;
  float l[C2_];
  float mx = -3.4e38f;
#pragma unroll
  for (int c = 0; c < C2_; ++c) {
    l[c] = agg[(size_t)n * C2_ + c] + bias[c];
    mx = fmaxf(mx, l[c]);
  }
  float s = 0.0f;
#pragma unroll
  for (int c = 0; c < C2_; ++c) s += __expf(l[c] - mx);
  float lse = mx + __logf(s);
#pragma unroll
  for (int c = 0; c < C2_; ++c) out[(size_t)n * C2_ + c] = l[c] - lse;
}

// ---------------- launch ----------------
extern "C" void kernel_launch(void* const* d_in, const int* in_sizes, int n_in,
                              void* d_out, int out_size, void* d_ws, size_t ws_size,
                              hipStream_t stream) {
  const float*     x    = (const float*)d_in[0];
  const long long* pi   = (const long long*)d_in[1];  // int64 [3,P]
  const float*     W1   = (const float*)d_in[2];      // [512,64]
  const float*     att1 = (const float*)d_in[3];      // [3,8,8]
  const float*     b1   = (const float*)d_in[4];      // [64]
  const float*     W2   = (const float*)d_in[5];      // [64,7]
  const float*     att2 = (const float*)d_in[6];      // [3,1,7]
  const float*     b2   = (const float*)d_in[7];      // [7]
  float*           out  = (float*)d_out;

  // workspace layout (floats first, then 32B-aligned bf16 fragment buffers)
  float* ws   = (float*)d_ws;
  float* xp1  = ws;                                    // N*64
  float* hbuf = xp1  + (size_t)NNODES * HC1;           // N*64 (agg1 -> elu in place)
  float* e1   = hbuf + (size_t)NNODES * HC1;           // P*8
  float* mx1  = e1   + (size_t)NPATH * H1_;            // N*8
  float* dn1  = mx1  + (size_t)NNODES * H1_;           // N*8
  float* xp2  = dn1  + (size_t)NNODES * H1_;           // N*7
  float* e2   = xp2  + (size_t)NNODES * C2_;           // P
  float* mx2  = e2   + (size_t)NPATH;                  // N
  float* dn2  = mx2  + (size_t)NNODES;                 // N
  float* agg2 = dn2  + (size_t)NNODES;                 // N*7
  // bf16 fragment buffers (32B aligned: offsets are multiples of 8 floats)
  size_t fend = ((size_t)(agg2 + (size_t)NNODES * C2_ - ws) + 7) & ~(size_t)7;
  __bf16* Bp1 = (__bf16*)(ws + fend);                  // 16*4*32*16 = 32768 bf16
  __bf16* Bp2 = Bp1 + 16 * 4 * 32 * 16;                // 2*1*32*16  = 1024 bf16

  const float NEG_INF = -__builtin_inff();
  auto cdiv = [](long long a, long long b) { return (unsigned)((a + b - 1) / b); };

  constexpr int MTILES  = (NNODES + 15) / 16;          // 6250
  constexpr int GEMM_WV = 4;                           // waves per block

  // ---------- layer 1 ----------
  pack_B<16, 4><<<cdiv(16 * 4 * 32, 256), 256, 0, stream>>>(W1, Bp1, HC1);
  wmma_gemm_bf16<16, 4><<<cdiv(MTILES, GEMM_WV), GEMM_WV * 32, 0, stream>>>(x, Bp1, xp1, NNODES, HC1);
  fill_f32<<<cdiv((long long)NNODES * H1_, 256), 256, 0, stream>>>(mx1, (size_t)NNODES * H1_, NEG_INF);
  fill_f32<<<cdiv((long long)NNODES * H1_, 256), 256, 0, stream>>>(dn1, (size_t)NNODES * H1_, 0.0f);
  fill_f32<<<cdiv((long long)NNODES * HC1, 256), 256, 0, stream>>>(hbuf, (size_t)NNODES * HC1, 0.0f);
  path_logits_max<H1_, C1_><<<NPATH / 256, 256, 0, stream>>>(xp1, pi, att1, e1, mx1);
  path_exp_sum<H1_><<<NPATH / 256, 256, 0, stream>>>(pi, mx1, e1, dn1);
  path_scatter<H1_, C1_><<<NPATH / 256, 256, 0, stream>>>(xp1, pi, e1, dn1, hbuf);
  bias_elu<<<cdiv((long long)NNODES * HC1, 256), 256, 0, stream>>>(hbuf, b1, HC1, (long long)NNODES * HC1);

  // ---------- layer 2 ----------
  pack_B<2, 1><<<1, 64, 0, stream>>>(W2, Bp2, C2_);
  wmma_gemm_bf16<2, 1><<<cdiv(MTILES, GEMM_WV), GEMM_WV * 32, 0, stream>>>(hbuf, Bp2, xp2, NNODES, C2_);
  fill_f32<<<cdiv(NNODES, 256), 256, 0, stream>>>(mx2, (size_t)NNODES, NEG_INF);
  fill_f32<<<cdiv(NNODES, 256), 256, 0, stream>>>(dn2, (size_t)NNODES, 0.0f);
  fill_f32<<<cdiv((long long)NNODES * C2_, 256), 256, 0, stream>>>(agg2, (size_t)NNODES * C2_, 0.0f);
  path_logits_max<H2_, C2_><<<NPATH / 256, 256, 0, stream>>>(xp2, pi, att2, e2, mx2);
  path_exp_sum<H2_><<<NPATH / 256, 256, 0, stream>>>(pi, mx2, e2, dn2);
  path_scatter<H2_, C2_><<<NPATH / 256, 256, 0, stream>>>(xp2, pi, e2, dn2, agg2);
  bias_logsoftmax7<<<cdiv(NNODES, 256), 256, 0, stream>>>(agg2, b2, out);
}
